// AttentionDecoder_81758997446680
// MI455X (gfx1250) — matrix operand order repaired and checked by reading, not compiled, not run on hardware
//
#include <hip/hip_runtime.h>
#include <math.h>

// ---------------------------------------------------------------------------
// Problem constants (from reference)
// ---------------------------------------------------------------------------
static constexpr int B_   = 64;
static constexpr int HW_  = 14;
static constexpr int P_   = HW_ * HW_;   // 196 pixels
static constexpr int E_   = 2048;        // encoder dim
static constexpr int A_   = 512;         // attention dim
static constexpr int D_   = 512;         // decoder dim
static constexpr int EMB_ = 512;         // embedding dim
static constexpr int V_   = 10000;       // vocab
static constexpr int L_   = 52;
static constexpr int T_   = L_ - 1;      // 51 steps
static constexpr int XIN_ = EMB_ + E_;   // 2560

// ---------------------------------------------------------------------------
// CDNA5 WMMA types
// ---------------------------------------------------------------------------
typedef __attribute__((ext_vector_type(16))) __bf16 v16bf;
typedef __attribute__((ext_vector_type(8)))  float  v8f;

static __device__ __forceinline__ v8f wmma_bf16(v16bf a, v16bf b, v8f c) {
  // D = A(16x32 bf16) * B(32x16 bf16) + C(16x16 f32)
  return __builtin_amdgcn_wmma_f32_16x16x32_bf16(
      /*neg_a=*/false, a, /*neg_b=*/false, b,
      /*c_mod=*/(short)0, c, /*reuse_a=*/false, /*reuse_b=*/false);
}

// round-to-nearest-even f32 -> bf16 (bit-level; no __bf16 arithmetic needed)
static __device__ __forceinline__ unsigned short f2bf(float f) {
  unsigned int u = __float_as_uint(f);
  unsigned int lsb = (u >> 16) & 1u;
  u += 0x7fffu + lsb;
  return (unsigned short)(u >> 16);
}
static __device__ __forceinline__ unsigned int pack2(float lo, float hi) {
  return (unsigned int)f2bf(lo) | ((unsigned int)f2bf(hi) << 16);
}
static __device__ __forceinline__ float sigf(float x) {
  return 1.0f / (1.0f + __expf(-x));
}

// ---------------------------------------------------------------------------
// Generic GEMM:  C[M,N] = act( A[M,K](f32) * W[N,K]^T(bf16) + bias ) (+C)
//   flags bit0: sigmoid epilogue
//   flags bit1: accumulate into existing C
//   flags bit2: row-mask: zero row b when !(t < lens[b]-1)
// M % 64 == 0 (uses: 64 and 12544). K % 64 == 0 (512/2048/2560).
// N may be ragged (fc: 10000) -> clamped loads + guarded stores.
// Block: 256 threads = 8 waves; block tile 64x128; BK=64.
// Wave tile 16x64: 4 accumulators -> 8 v_wmma per barrier interval,
// A-fragment reuse x4.
// ---------------------------------------------------------------------------
__global__ __launch_bounds__(256)
void gemm_bf16_wmma(const float* __restrict__ Amat, int lda,
                    const unsigned short* __restrict__ Wmat, int ldw,
                    const float* __restrict__ bias,
                    float* __restrict__ C, int ldc,
                    int M, int N, int K,
                    int flags, const int* __restrict__ lens, int t)
{
  __shared__ __align__(16) unsigned short As[64][72];   // 64 rows  x 64 k (+pad)
  __shared__ __align__(16) unsigned short Ws[128][72];  // 128 rows x 64 k (+pad)

  const int bm   = blockIdx.y * 64;
  const int bn   = blockIdx.x * 128;
  const int tid  = threadIdx.x;
  const int lane = tid & 31;
  const int wave = tid >> 5;
  const int wm   = wave >> 1;       // 0..3   (16-row strip)
  const int wn   = wave & 1;        // 0..1   (64-col strip)

  v8f acc[4] = {v8f{}, v8f{}, v8f{}, v8f{}};

  // A-tile loader mapping: 64 rows x 64 k fp32, 16 floats/thread
  const int arow = tid >> 2;          // 0..63
  const int acol = (tid & 3) * 16;    // 0,16,32,48
  // W-tile loader mapping: 128 rows x 64 k bf16, 32 bf16/thread
  const int wlrow = tid >> 1;         // 0..127
  const int wlcol = (tid & 1) * 32;   // 0,32

  // fragment mapping (CDNA5 ISA 7.12.2 half-lane K split)
  const int lrow  = lane & 15;
  const int lhi   = lane >> 4;
  const int khalf = lhi * 8;          // A: K 0..7 / 8..15 (+16 in hi regs)
  const int kbase = lhi * 16;         // unused (B loads use khalf, same k order as A)

  for (int k0 = 0; k0 < K; k0 += 64) {
    // ---- load A tile (fp32 -> bf16) ----
    const float* ap = Amat + (size_t)(bm + arow) * lda + (k0 + acol);
    float4 a0 = *(const float4*)(ap);
    float4 a1 = *(const float4*)(ap + 4);
    float4 a2 = *(const float4*)(ap + 8);
    float4 a3 = *(const float4*)(ap + 12);
    // ---- load W tile (bf16, N-row clamped for ragged N) ----
    int wrow = bn + wlrow; if (wrow >= N) wrow = N - 1;
    const unsigned short* wp = Wmat + (size_t)wrow * ldw + (k0 + wlcol);
    uint4 w0 = *(const uint4*)(wp);
    uint4 w1 = *(const uint4*)(wp + 8);
    uint4 w2 = *(const uint4*)(wp + 16);
    uint4 w3 = *(const uint4*)(wp + 24);

    if (k0 + 64 < K) {                       // -> global_prefetch_b8
      __builtin_prefetch((const void*)(ap + 64), 0, 1);
      __builtin_prefetch((const void*)(wp + 64), 0, 1);
    }

    uint4 u0, u1;
    u0.x = pack2(a0.x, a0.y); u0.y = pack2(a0.z, a0.w);
    u0.z = pack2(a1.x, a1.y); u0.w = pack2(a1.z, a1.w);
    u1.x = pack2(a2.x, a2.y); u1.y = pack2(a2.z, a2.w);
    u1.z = pack2(a3.x, a3.y); u1.w = pack2(a3.z, a3.w);
    *(uint4*)&As[arow][acol]     = u0;
    *(uint4*)&As[arow][acol + 8] = u1;
    *(uint4*)&Ws[wlrow][wlcol]      = w0;
    *(uint4*)&Ws[wlrow][wlcol + 8]  = w1;
    *(uint4*)&Ws[wlrow][wlcol + 16] = w2;
    *(uint4*)&Ws[wlrow][wlcol + 24] = w3;
    __syncthreads();

#pragma unroll
    for (int ks = 0; ks < 2; ++ks) {
      const int ko = ks * 32;
      union { v16bf v; uint4 q[2]; } fa, fb;
      fa.q[0] = *(const uint4*)&As[wm * 16 + lrow][ko + khalf];
      fa.q[1] = *(const uint4*)&As[wm * 16 + lrow][ko + khalf + 16];
#pragma unroll
      for (int nt = 0; nt < 4; ++nt) {
        fb.q[0] = *(const uint4*)&Ws[wn * 64 + nt * 16 + lrow][ko + khalf];
        fb.q[1] = *(const uint4*)&Ws[wn * 64 + nt * 16 + lrow][ko + khalf + 16];
        acc[nt] = wmma_bf16(fa.v, fb.v, acc[nt]);
      }
    }
    __syncthreads();
  }

  // ---- epilogue (C/D layout: VGPR r -> row r + 8*(lane>>4), col = lane&15) ----
  const int col   = lane & 15;
  const int rbase = (lane >> 4) * 8;
  const bool do_sig  = (flags & 1) != 0;
  const bool do_acc  = (flags & 2) != 0;
  const bool do_mask = (flags & 4) != 0;
#pragma unroll
  for (int r = 0; r < 8; ++r) {
    const int gm = bm + wm * 16 + rbase + r;
#pragma unroll
    for (int nt = 0; nt < 4; ++nt) {
      const int gn = bn + wn * 64 + nt * 16 + col;
      if (gn < N) {
        float v = acc[nt][r];
        if (bias)   v += bias[gn];
        if (do_sig) v = sigf(v);
        if (do_acc) v += C[(size_t)gm * ldc + gn];
        if (do_mask && !(t < lens[gm] - 1)) v = 0.0f;
        C[(size_t)gm * ldc + gn] = v;
      }
    }
  }
}

// ---------------------------------------------------------------------------
// fp32 -> bf16 weight conversion
// ---------------------------------------------------------------------------
__global__ __launch_bounds__(256)
void k_convert(const float* __restrict__ src, unsigned short* __restrict__ dst, int n) {
  int i = blockIdx.x * 256 + threadIdx.x;
  if (i < n) dst[i] = f2bf(src[i]);
}

// ---------------------------------------------------------------------------
// mean over pixels: mean[b,e] = (1/P) * sum_p enc[b,p,e]
// ---------------------------------------------------------------------------
__global__ __launch_bounds__(256)
void k_mean(const float* __restrict__ enc, float* __restrict__ mean) {
  int i = blockIdx.x * 256 + threadIdx.x;         // over B*E
  if (i >= B_ * E_) return;
  int b = i / E_, e = i % E_;
  const float* p = enc + (size_t)b * P_ * E_ + e;
  float s = 0.f;
  for (int k = 0; k < P_; ++k) s += p[(size_t)k * E_];
  mean[i] = s * (1.0f / (float)P_);
}

// ---------------------------------------------------------------------------
// Fused attention step: score -> softmax over P -> awe, one block per batch b.
// score[p] = relu(att_enc[b,p,:] + att_dec[b,:]) . fullW + fullb
// awe[b,e] = sum_p alpha[p] * enc[b,p,e];  alpha written (masked) to d_out.
// ---------------------------------------------------------------------------
__global__ __launch_bounds__(256)
void k_attn(const float* __restrict__ att_enc, const float* __restrict__ att_dec,
            const float* __restrict__ fullW, const float* __restrict__ fullb,
            const float* __restrict__ enc,
            float* __restrict__ awe, float* __restrict__ alpha_out,
            const int* __restrict__ lens, int t)
{
  __shared__ float adec[A_];
  __shared__ float fw[A_];
  __shared__ float sc[P_];
  __shared__ float red[256];

  const int b = blockIdx.x, tid = threadIdx.x;
  adec[tid]       = att_dec[b * A_ + tid];
  adec[tid + 256] = att_dec[b * A_ + tid + 256];
  fw[tid]         = fullW[tid];
  fw[tid + 256]   = fullW[tid + 256];
  __syncthreads();

  float s = 0.f;
  if (tid < P_) {
    const float* ae = att_enc + ((size_t)b * P_ + tid) * A_;
    for (int a = 0; a < A_; ++a) {
      float x = fmaxf(ae[a] + adec[a], 0.0f);
      s = fmaf(x, fw[a], s);
    }
    s += fullb[0];
  }
  // max reduction
  red[tid] = (tid < P_) ? s : -3.0e38f;
  __syncthreads();
  for (int st = 128; st > 0; st >>= 1) {
    if (tid < st) red[tid] = fmaxf(red[tid], red[tid + st]);
    __syncthreads();
  }
  const float mx = red[0];
  __syncthreads();
  // sum reduction
  float ex = (tid < P_) ? __expf(s - mx) : 0.f;
  red[tid] = ex;
  __syncthreads();
  for (int st = 128; st > 0; st >>= 1) {
    if (tid < st) red[tid] += red[tid + st];
    __syncthreads();
  }
  const float denom = red[0];
  const float al = ex / denom;
  if (tid < P_) {
    sc[tid] = al;
    const bool m = t < lens[b] - 1;
    alpha_out[((size_t)b * T_ + t) * P_ + tid] = m ? al : 0.0f;
  }
  __syncthreads();

  // awe: each thread handles 8 strided e-columns (coalesced across lanes)
#pragma unroll
  for (int j = 0; j < E_ / 256; ++j) {
    const int e = tid + j * 256;
    float acc = 0.f;
    const float* ep = enc + (size_t)b * P_ * E_ + e;
    for (int p = 0; p < P_; ++p) acc = fmaf(sc[p], ep[(size_t)p * E_], acc);
    awe[b * E_ + e] = acc;
  }
}

// ---------------------------------------------------------------------------
// x_in = [ emb_W[caption[b,t]] , gate * awe ]   (B x 2560, fp32)
// ---------------------------------------------------------------------------
__global__ __launch_bounds__(256)
void k_xin(const float* __restrict__ embW, const int* __restrict__ caps,
           const float* __restrict__ gate, const float* __restrict__ awe,
           float* __restrict__ xin, int t)
{
  int i = blockIdx.x * 256 + threadIdx.x;         // over B*XIN_
  if (i >= B_ * XIN_) return;
  int b = i / XIN_, col = i % XIN_;
  float v;
  if (col < EMB_) {
    int tok = caps[b * L_ + t];
    v = embW[(size_t)tok * EMB_ + col];
  } else {
    int e = col - EMB_;
    v = gate[b * E_ + e] * awe[b * E_ + e];
  }
  xin[i] = v;
}

// ---------------------------------------------------------------------------
// LSTM pointwise + masked state update
// ---------------------------------------------------------------------------
__global__ __launch_bounds__(256)
void k_lstm(const float* __restrict__ gates,
            float* __restrict__ h, float* __restrict__ c,
            float* __restrict__ hnew,
            const int* __restrict__ lens, int t)
{
  int i = blockIdx.x * 256 + threadIdx.x;         // over B*D
  if (i >= B_ * D_) return;
  int b = i / D_, d = i % D_;
  const float* g = gates + (size_t)b * (4 * D_);
  float gi = g[d], gf = g[d + D_], gg = g[d + 2 * D_], go = g[d + 3 * D_];
  float cn = sigf(gf) * c[i] + sigf(gi) * tanhf(gg);
  float hn = sigf(go) * tanhf(cn);
  hnew[i] = hn;
  if (t < lens[b] - 1) { h[i] = hn; c[i] = cn; }
}

// ---------------------------------------------------------------------------
// Host orchestration
// ---------------------------------------------------------------------------
extern "C" void kernel_launch(void* const* d_in, const int* in_sizes, int n_in,
                              void* d_out, int out_size, void* d_ws, size_t ws_size,
                              hipStream_t stream)
{
  (void)in_sizes; (void)n_in; (void)out_size; (void)ws_size;
  const float* enc       = (const float*)d_in[0];   // (B,HW,HW,E) == (B,P,E)
  const int*   captions  = (const int*)d_in[1];     // (B,L)
  const int*   lens      = (const int*)d_in[2];     // (B)
  const float* emb_W     = (const float*)d_in[3];   // (V,EMB)
  const float* enc_att_W = (const float*)d_in[4];   // (A,E)
  const float* enc_att_b = (const float*)d_in[5];
  const float* dec_att_W = (const float*)d_in[6];   // (A,D)
  const float* dec_att_b = (const float*)d_in[7];
  const float* full_W    = (const float*)d_in[8];   // (1,A)
  const float* full_b    = (const float*)d_in[9];
  const float* h_lin_W   = (const float*)d_in[10];  // (D,E)
  const float* h_lin_b   = (const float*)d_in[11];
  const float* c_lin_W   = (const float*)d_in[12];  // (D,E)
  const float* c_lin_b   = (const float*)d_in[13];
  const float* f_beta_W  = (const float*)d_in[14];  // (E,D)
  const float* f_beta_b  = (const float*)d_in[15];
  const float* lstm_Wih  = (const float*)d_in[16];  // (4D, XIN)
  const float* lstm_Whh  = (const float*)d_in[17];  // (4D, D)
  const float* lstm_bih  = (const float*)d_in[18];
  const float* lstm_bhh  = (const float*)d_in[19];
  const float* fc_W      = (const float*)d_in[20];  // (V,D)
  const float* fc_b      = (const float*)d_in[21];

  float* out = (float*)d_out;
  float* alpha_out = out + (size_t)B_ * T_ * V_;    // attention weights region

  // ---- workspace carve-up ----
  size_t off = 0;
  auto alloc_bf = [&](size_t n) {
    unsigned short* p = (unsigned short*)((char*)d_ws + off);
    off += ((n * 2 + 255) / 256) * 256; return p;
  };
  auto alloc_f = [&](size_t n) {
    float* p = (float*)((char*)d_ws + off);
    off += ((n * 4 + 255) / 256) * 256; return p;
  };
  unsigned short* enc_att_bf = alloc_bf((size_t)A_ * E_);
  unsigned short* dec_att_bf = alloc_bf((size_t)A_ * D_);
  unsigned short* h_lin_bf   = alloc_bf((size_t)D_ * E_);
  unsigned short* c_lin_bf   = alloc_bf((size_t)D_ * E_);
  unsigned short* f_beta_bf  = alloc_bf((size_t)E_ * D_);
  unsigned short* Wih_bf     = alloc_bf((size_t)4 * D_ * XIN_);
  unsigned short* Whh_bf     = alloc_bf((size_t)4 * D_ * D_);
  unsigned short* fc_bf      = alloc_bf((size_t)V_ * D_);
  float* att_enc = alloc_f((size_t)B_ * P_ * A_);
  float* meanenc = alloc_f((size_t)B_ * E_);
  float* h       = alloc_f((size_t)B_ * D_);
  float* c       = alloc_f((size_t)B_ * D_);
  float* hnew    = alloc_f((size_t)B_ * D_);
  float* att_dec = alloc_f((size_t)B_ * A_);
  float* awe     = alloc_f((size_t)B_ * E_);
  float* gate    = alloc_f((size_t)B_ * E_);
  float* xin     = alloc_f((size_t)B_ * XIN_);
  float* gates   = alloc_f((size_t)B_ * 4 * D_);

  auto conv = [&](const float* s, unsigned short* d, size_t n) {
    k_convert<<<dim3((unsigned)((n + 255) / 256)), 256, 0, stream>>>(s, d, (int)n);
  };
  conv(enc_att_W, enc_att_bf, (size_t)A_ * E_);
  conv(dec_att_W, dec_att_bf, (size_t)A_ * D_);
  conv(h_lin_W,   h_lin_bf,   (size_t)D_ * E_);
  conv(c_lin_W,   c_lin_bf,   (size_t)D_ * E_);
  conv(f_beta_W,  f_beta_bf,  (size_t)E_ * D_);
  conv(lstm_Wih,  Wih_bf,     (size_t)4 * D_ * XIN_);
  conv(lstm_Whh,  Whh_bf,     (size_t)4 * D_ * D_);
  conv(fc_W,      fc_bf,      (size_t)V_ * D_);

  auto gemm = [&](const float* Amat, int lda, const unsigned short* Wm, int ldw,
                  const float* bias, float* C, int ldc, int M, int N, int K,
                  int flags, int t) {
    dim3 grid((N + 127) / 128, (M + 63) / 64);
    gemm_bf16_wmma<<<grid, 256, 0, stream>>>(Amat, lda, Wm, ldw, bias, C, ldc,
                                             M, N, K, flags, lens, t);
  };

  // ---- precompute ----
  // att_enc = enc @ enc_att_W^T + b   (M=12544, N=512, K=2048)
  gemm(enc, E_, enc_att_bf, E_, enc_att_b, att_enc, A_, B_ * P_, A_, E_, 0, 0);
  // mean over pixels
  k_mean<<<dim3((B_ * E_) / 256), 256, 0, stream>>>(enc, meanenc);
  // h0 / c0
  gemm(meanenc, E_, h_lin_bf, E_, h_lin_b, h, D_, B_, D_, E_, 0, 0);
  gemm(meanenc, E_, c_lin_bf, E_, c_lin_b, c, D_, B_, D_, E_, 0, 0);

  // ---- sequential decode ----
  for (int t = 0; t < T_; ++t) {
    // att_dec = h @ dec_att_W^T + b
    gemm(h, D_, dec_att_bf, D_, dec_att_b, att_dec, A_, B_, A_, D_, 0, t);
    // attention: score -> softmax -> awe (+ masked alpha to d_out)
    k_attn<<<dim3(B_), 256, 0, stream>>>(att_enc, att_dec, full_W, full_b,
                                         enc, awe, alpha_out, lens, t);
    // gate = sigmoid(h @ f_beta_W^T + b)
    gemm(h, D_, f_beta_bf, D_, f_beta_b, gate, E_, B_, E_, D_, 1, t);
    // x_in = [emb, gate*awe]
    k_xin<<<dim3((B_ * XIN_) / 256), 256, 0, stream>>>(emb_W, captions, gate,
                                                       awe, xin, t);
    // gates = x_in @ Wih^T + bih
    gemm(xin, XIN_, Wih_bf, XIN_, lstm_bih, gates, 4 * D_, B_, 4 * D_, XIN_, 0, t);
    // gates += h @ Whh^T + bhh
    gemm(h, D_, Whh_bf, D_, lstm_bhh, gates, 4 * D_, B_, 4 * D_, D_, 2, t);
    // LSTM pointwise + masked state update
    k_lstm<<<dim3((B_ * D_) / 256), 256, 0, stream>>>(gates, h, c, hnew, lens, t);
    // preds = hnew @ fc_W^T + fc_b, masked, strided into d_out[(b,t,:)]
    gemm(hnew, D_, fc_bf, D_, fc_b, out + (size_t)t * V_, T_ * V_,
         B_, V_, D_, /*flags=*/4, t);
  }
}
